// AttentionSemantic_27358941675744
// MI455X (gfx1250) — compile-verified, matrix operand order checked by reading, not run on hardware
//
#include <hip/hip_runtime.h>
#include <hip/hip_bf16.h>
#include <math.h>

#define B_DIM 64
#define L_DIM 2048
#define E_DIM 1024
#define A_DIM 512

#define ROWS   128            // L-rows per workgroup in the main kernel
#define KPAIRS (E_DIM / 2)    // 512 packed-bf16 pairs per row
#define ASTRIDE 516           // 512 + 4 dword pad -> conflict-free banks, keeps 16B align

typedef __attribute__((ext_vector_type(16))) __bf16 v16bf;
typedef __attribute__((ext_vector_type(8)))  float  v8f;
typedef __attribute__((ext_vector_type(8)))  unsigned int v8u;
typedef __attribute__((ext_vector_type(4)))  float  f32x4;   // clang-native, NT-load friendly

__device__ __forceinline__ unsigned int pack2bf16(float a, float b) {
    unsigned int ua = __builtin_bit_cast(unsigned int, a);
    unsigned int ub = __builtin_bit_cast(unsigned int, b);
    ua = (ua + 0x7FFFu + ((ua >> 16) & 1u)) >> 16;   // RNE f32 -> bf16
    ub = (ub + 0x7FFFu + ((ub >> 16) & 1u)) >> 16;
    return (ua & 0xFFFFu) | (ub << 16);
}

// branch-free tanh: 1 - 2/(e^{2x}+1); e->inf => 1, e->0 => -1 (no NaN/divergence)
__device__ __forceinline__ float fast_tanh(float x) {
    float e = __expf(2.0f * x);
    return 1.0f - 2.0f / (e + 1.0f);
}

// load one 16x32-bf16 WMMA fragment (8 dwords: base+0..3, base+8..11) from LDS
__device__ __forceinline__ v16bf load_frag(const unsigned int* base, int dw) {
    const uint4* p = (const uint4*)(base + dw);
    uint4 x0 = p[0], x1 = p[2];
    v8u u;
    u[0] = x0.x; u[1] = x0.y; u[2] = x0.z; u[3] = x0.w;
    u[4] = x1.x; u[5] = x1.y; u[6] = x1.z; u[7] = x1.w;
    return __builtin_bit_cast(v16bf, u);
}

// ---------------------------------------------------------------------------
// K0: dec_proj[64,512] = dec_hidden[64,1024] @ W_dec[1024,512] + b_dec
// ---------------------------------------------------------------------------
__global__ __launch_bounds__(256)
void dec_proj_kernel(const float* __restrict__ dec, const float* __restrict__ Wd,
                     const float* __restrict__ bd, float* __restrict__ out) {
    int o = blockIdx.x * 256 + threadIdx.x;          // 32768 outputs
    int b = o >> 9, c = o & 511;
    float acc = bd[c];
    const float* dh = dec + b * E_DIM;
    for (int k = 0; k < E_DIM; ++k) acc = fmaf(dh[k], Wd[k * A_DIM + c], acc);
    out[o] = acc;
}

// ---------------------------------------------------------------------------
// K1: fused enc-projection GEMM (bf16 WMMA, f32 accum) + tanh + score dot.
// One WG = 8 waves handles 128 L-rows x all 512 att columns of one batch.
// ---------------------------------------------------------------------------
__global__ __launch_bounds__(256)
void enc_score_kernel(const float* __restrict__ X, const float* __restrict__ We,
                      const float* __restrict__ be, const float* __restrict__ dproj,
                      const float* __restrict__ Wf, const float* __restrict__ bfull,
                      float* __restrict__ scores_raw) {
    __shared__ __align__(16) unsigned int As[ROWS * ASTRIDE];   // 258 KB
    __shared__ __align__(16) unsigned int Bs[16   * ASTRIDE];   //  33 KB

    const int tid  = threadIdx.x;
    const int wave = tid >> 5;
    const int lane = tid & 31;
    const int half = lane >> 4;       // 0: lanes 0-15, 1: lanes 16-31
    const int l16  = lane & 15;
    const int b    = blockIdx.y;
    const int l0   = blockIdx.x * ROWS;
    const int wrow = wave * 16;       // this wave's 16-row tile inside the block

    // ---- stage A tile: 128 x 1024 f32 -> packed bf16 pairs in LDS ----
    // X is streamed once and won't be revisited by this pass: non-temporal.
    {
        const f32x4* src = (const f32x4*)(X + ((size_t)b * L_DIM + l0) * E_DIM);
        for (int q = tid; q < ROWS * (E_DIM / 4); q += 256) {
            int row = q >> 8;                 // 256 float4 per row
            int p4  = q & 255;
            f32x4 v = __builtin_nontemporal_load(src + (size_t)row * 256 + p4);
            uint2 pk;
            pk.x = pack2bf16(v.x, v.y);
            pk.y = pack2bf16(v.z, v.w);
            *(uint2*)(As + row * ASTRIDE + 2 * p4) = pk;   // 8B-aligned
        }
    }

    float part[8];
#pragma unroll
    for (int r = 0; r < 8; ++r) part[r] = 0.f;

    for (int ct = 0; ct < A_DIM / 16; ++ct) {
        __syncthreads();  // A visible (ct==0) / previous Bs consumed (ct>0)

        // ---- stage W_enc column tile: 1024 x 16 f32 -> bf16 pairs ----
        for (int q = tid; q < 16 * KPAIRS; q += 256) {
            int col = q & 15;
            int kp  = q >> 4;
            float x0 = We[(2 * kp)     * A_DIM + ct * 16 + col];
            float x1 = We[(2 * kp + 1) * A_DIM + ct * 16 + col];
            Bs[col * ASTRIDE + kp] = pack2bf16(x0, x1);
        }
        __syncthreads();

        // ---- 16x16 output tile: 32 x v_wmma_f32_16x16x32_bf16 over K=1024,
        //      split into 4 independent accumulator chains to hide XDL latency
        v8f acc0 = {}, acc1 = {}, acc2 = {}, acc3 = {};
        const unsigned int* arow = As + (wrow + l16) * ASTRIDE;
        const unsigned int* bcol = Bs + l16 * ASTRIDE;
        const int hb = half * 4;              // lane-half k-pair offset
#pragma unroll 2
        for (int ks = 0; ks < E_DIM / 32; ks += 4) {
            // ISA 16-bit A/B layout: lanes 0-15 K={0..7,16..23}, lanes 16-31 +8
            int d0 = (ks + 0) * 16 + hb;
            int d1 = (ks + 1) * 16 + hb;
            int d2 = (ks + 2) * 16 + hb;
            int d3 = (ks + 3) * 16 + hb;
            acc0 = __builtin_amdgcn_wmma_f32_16x16x32_bf16(
                false, load_frag(arow, d0), false, load_frag(bcol, d0),
                (short)0, acc0, false, false);
            acc1 = __builtin_amdgcn_wmma_f32_16x16x32_bf16(
                false, load_frag(arow, d1), false, load_frag(bcol, d1),
                (short)0, acc1, false, false);
            acc2 = __builtin_amdgcn_wmma_f32_16x16x32_bf16(
                false, load_frag(arow, d2), false, load_frag(bcol, d2),
                (short)0, acc2, false, false);
            acc3 = __builtin_amdgcn_wmma_f32_16x16x32_bf16(
                false, load_frag(arow, d3), false, load_frag(bcol, d3),
                (short)0, acc3, false, false);
        }
        v8f acc = (acc0 + acc1) + (acc2 + acc3);

        // ---- fused epilogue: tanh(enc+dec+b_enc) * W_full, per C element ----
        // C layout: VGPR r -> row (r + half*8), lane&15 -> col within tile.
        int   col = ct * 16 + l16;
        float dpv = dproj[b * A_DIM + col] + be[col];
        float wfv = Wf[col];
#pragma unroll
        for (int r = 0; r < 8; ++r) {
            float t = fast_tanh(acc[r] + dpv);
            part[r] = fmaf(t, wfv, part[r]);
        }
    }

    // reduce the 16 lanes (columns) that share each row
#pragma unroll
    for (int m = 1; m < 16; m <<= 1)
#pragma unroll
        for (int r = 0; r < 8; ++r)
            part[r] += __shfl_xor(part[r], m, 32);

    if (l16 == 0) {
        float bf0 = bfull[0];
        int rowbase = l0 + wrow + half * 8;
#pragma unroll
        for (int r = 0; r < 8; ++r)
            scores_raw[b * L_DIM + rowbase + r] = part[r] + bf0;
    }
}

// ---------------------------------------------------------------------------
// K2: softmax over L per batch row
// ---------------------------------------------------------------------------
__global__ __launch_bounds__(256)
void softmax_kernel(const float* __restrict__ sraw, float* __restrict__ p) {
    __shared__ float red[256];
    int b = blockIdx.x, tid = threadIdx.x;
    const float* s = sraw + b * L_DIM;

    float mx = -__FLT_MAX__;
    for (int l = tid; l < L_DIM; l += 256) mx = fmaxf(mx, s[l]);
    red[tid] = mx; __syncthreads();
    for (int st = 128; st > 0; st >>= 1) {
        if (tid < st) red[tid] = fmaxf(red[tid], red[tid + st]);
        __syncthreads();
    }
    mx = red[0]; __syncthreads();

    float sum = 0.f;
    for (int l = tid; l < L_DIM; l += 256) sum += __expf(s[l] - mx);
    red[tid] = sum; __syncthreads();
    for (int st = 128; st > 0; st >>= 1) {
        if (tid < st) red[tid] += red[tid + st];
        __syncthreads();
    }
    float inv = 1.f / red[0];
    for (int l = tid; l < L_DIM; l += 256) p[b * L_DIM + l] = __expf(s[l] - mx) * inv;
}

// ---------------------------------------------------------------------------
// K3: y[b,e] = sum_l p[b,l] * X[b,l,e]   (p^T X), coalesced over e
// ---------------------------------------------------------------------------
__global__ __launch_bounds__(256)
void wsum_kernel(const float* __restrict__ X, const float* __restrict__ p,
                 float* __restrict__ y) {
    __shared__ float pl[L_DIM];                       // 8 KB
    int b = blockIdx.y;
    int e = blockIdx.x * 256 + threadIdx.x;
    for (int l = threadIdx.x; l < L_DIM; l += 256) pl[l] = p[b * L_DIM + l];
    __syncthreads();
    const float* xb = X + (size_t)b * L_DIM * E_DIM + e;
    float acc = 0.f;
    for (int l = 0; l < L_DIM; ++l)
        acc = fmaf(pl[l], __builtin_nontemporal_load(xb + (size_t)l * E_DIM), acc);
    y[b * E_DIM + e] = acc;
}

// ---------------------------------------------------------------------------
// K4: att_output[64,512] = y[64,1024] @ W_enc + b_enc   (softmax sums to 1)
// ---------------------------------------------------------------------------
__global__ __launch_bounds__(256)
void out_proj_kernel(const float* __restrict__ y, const float* __restrict__ We,
                     const float* __restrict__ be, float* __restrict__ out) {
    int o = blockIdx.x * 256 + threadIdx.x;           // 32768 outputs
    int b = o >> 9, c = o & 511;
    float acc = be[c];
    const float* yb = y + b * E_DIM;
    for (int k = 0; k < E_DIM; ++k) acc = fmaf(yb[k], We[k * A_DIM + c], acc);
    out[o] = acc;
}

// ---------------------------------------------------------------------------
extern "C" void kernel_launch(void* const* d_in, const int* in_sizes, int n_in,
                              void* d_out, int out_size, void* d_ws, size_t ws_size,
                              hipStream_t stream) {
    const float* X   = (const float*)d_in[0];  // [64,2048,1024]
    const float* dec = (const float*)d_in[1];  // [64,1024]
    const float* We  = (const float*)d_in[2];  // [1024,512]
    const float* be  = (const float*)d_in[3];  // [512]
    const float* Wd  = (const float*)d_in[4];  // [1024,512]
    const float* bd  = (const float*)d_in[5];  // [512]
    const float* Wf  = (const float*)d_in[6];  // [512,1]
    const float* bf  = (const float*)d_in[7];  // [1]

    float* out_att = (float*)d_out;                 // [64,512]  first output
    float* out_p   = out_att + B_DIM * A_DIM;       // [64,2048] second output

    float* ws_dec  = (float*)d_ws;                  // [64,512]
    float* ws_sraw = ws_dec + B_DIM * A_DIM;        // [64,2048] raw scores
    float* ws_y    = ws_sraw + B_DIM * L_DIM;       // [64,1024] p^T X

    dec_proj_kernel<<<dim3(B_DIM * A_DIM / 256), dim3(256), 0, stream>>>(dec, Wd, bd, ws_dec);
    enc_score_kernel<<<dim3(L_DIM / ROWS, B_DIM), dim3(256), 0, stream>>>(
        X, We, be, ws_dec, Wf, bf, ws_sraw);
    softmax_kernel<<<dim3(B_DIM), dim3(256), 0, stream>>>(ws_sraw, out_p);
    wsum_kernel<<<dim3(E_DIM / 256, B_DIM), dim3(256), 0, stream>>>(X, out_p, ws_y);
    out_proj_kernel<<<dim3(B_DIM * A_DIM / 256), dim3(256), 0, stream>>>(ws_y, We, be, out_att);
}
